// GATEncoder_5050881540427
// MI455X (gfx1250) — matrix operand order repaired and checked by reading, not compiled, not run on hardware
//
#include <hip/hip_runtime.h>

typedef _Float16 h16 __attribute__((ext_vector_type(16)));
typedef _Float16 hp2 __attribute__((ext_vector_type(2)));
typedef float    f8  __attribute__((ext_vector_type(8)));

#define NEG_SLOPE 0.2f
#define ORD_NEG_INF 0x007FFFFFu   // order-encoding of -inf

__device__ __forceinline__ unsigned ford(float f) {
    unsigned u = __float_as_uint(f);
    return (u & 0x80000000u) ? ~u : (u | 0x80000000u);
}
__device__ __forceinline__ float funord(unsigned u) {
    return (u & 0x80000000u) ? __uint_as_float(u & 0x7FFFFFFFu)
                             : __uint_as_float(~u);
}
__device__ __forceinline__ void atomF(float* p, float v) {
    __hip_atomic_fetch_add(p, v, __ATOMIC_RELAXED, __HIP_MEMORY_SCOPE_AGENT);
}
__device__ __forceinline__ void edge_nodes(const int* __restrict__ ei, int E,
                                           int e, int& s, int& d) {
    if (e < E) { s = ei[e]; d = ei[E + e]; }
    else       { s = e - E; d = s; }
}

// ---------------- Layer 1: node transform + attention coefficients ----------
// One block (256 thr) per node; each wave (32 lanes) == one head.
__global__ void k_layer1_node(const float* __restrict__ x,
                              const float* __restrict__ W1,
                              const float* __restrict__ as1,
                              const float* __restrict__ ad1,
                              float* __restrict__ h1,
                              float* __restrict__ aS, float* __restrict__ aD) {
    int n = blockIdx.x, c = threadIdx.x;
    float x0 = x[n * 3 + 0], x1 = x[n * 3 + 1], x2 = x[n * 3 + 2];
    float h = x0 * W1[c] + x1 * W1[256 + c] + x2 * W1[512 + c];
    h1[(long)n * 256 + c] = h;
    float s = h * as1[c], d = h * ad1[c];
    #pragma unroll
    for (int off = 16; off; off >>= 1) {
        s += __shfl_xor(s, off, 32);
        d += __shfl_xor(d, off, 32);
    }
    if ((c & 31) == 0) {
        int head = c >> 5;
        aS[n * 8 + head] = s;
        aD[n * 8 + head] = d;
    }
}

__global__ void k_init1(float* __restrict__ out1, unsigned* __restrict__ emax,
                        float* __restrict__ denom, const float* __restrict__ b1,
                        int NC, int NH) {
    int i = blockIdx.x * blockDim.x + threadIdx.x;
    if (i < NC) out1[i] = b1[i & 255];
    if (i < NH) { emax[i] = ORD_NEG_INF; denom[i] = 0.f; }
}

__global__ void k_edge_max1(const int* __restrict__ ei, int E,
                            const float* __restrict__ aS, const float* __restrict__ aD,
                            unsigned* __restrict__ emax, int EE) {
    int gid = blockIdx.x * blockDim.x + threadIdx.x;
    if (gid >= EE * 8) return;
    int e = gid >> 3, h = gid & 7, s, d;
    edge_nodes(ei, E, e, s, d);
    float sc = aS[s * 8 + h] + aD[d * 8 + h];
    sc = sc > 0.f ? sc : NEG_SLOPE * sc;
    atomicMax(&emax[d * 8 + h], ford(sc));
}

__global__ void k_edge_sum1(const int* __restrict__ ei, int E,
                            const float* __restrict__ aS, const float* __restrict__ aD,
                            const unsigned* __restrict__ emax,
                            float* __restrict__ denom, int EE) {
    int gid = blockIdx.x * blockDim.x + threadIdx.x;
    if (gid >= EE * 8) return;
    int e = gid >> 3, h = gid & 7, s, d;
    edge_nodes(ei, E, e, s, d);
    float sc = aS[s * 8 + h] + aD[d * 8 + h];
    sc = sc > 0.f ? sc : NEG_SLOPE * sc;
    atomF(&denom[d * 8 + h], __expf(sc - funord(emax[d * 8 + h])));
}

// One block (256 thr) per edge: gather 256-ch row of h1[src], scatter to out1[dst].
__global__ void k_edge_agg1(const int* __restrict__ ei, int E,
                            const float* __restrict__ aS, const float* __restrict__ aD,
                            const unsigned* __restrict__ emax,
                            const float* __restrict__ denom,
                            const float* __restrict__ h1, float* __restrict__ out1) {
    int e = blockIdx.x, c = threadIdx.x, h = c >> 5, s, d;
    edge_nodes(ei, E, e, s, d);
    float sc = aS[s * 8 + h] + aD[d * 8 + h];
    sc = sc > 0.f ? sc : NEG_SLOPE * sc;
    float alpha = __expf(sc - funord(emax[d * 8 + h])) / (denom[d * 8 + h] + 1e-16f);
    atomF(&out1[(long)d * 256 + c], h1[(long)s * 256 + c] * alpha);
}

__global__ void k_elu_tohalf(const float* __restrict__ out1,
                             _Float16* __restrict__ hh, int NC) {
    int i = blockIdx.x * blockDim.x + threadIdx.x;
    if (i >= NC) return;
    float v = out1[i];
    v = v > 0.f ? v : (__expf(v) - 1.f);
    hh[i] = (_Float16)v;
}

// W2 [256,16] f32 -> W2t [16,256] f16 (transposed so B K-pairs are contiguous)
__global__ void k_w2t(const float* __restrict__ W2, _Float16* __restrict__ W2t) {
    int t = blockIdx.x * blockDim.x + threadIdx.x;
    if (t >= 4096) return;
    int nn = t >> 8, k = t & 255;
    W2t[nn * 256 + k] = (_Float16)W2[k * 16 + nn];
}

// ---------------- Layer 2 GEMM via WMMA: z[N,16] = hh[N,256] @ W2 ----------
__global__ void k_gemm2_wmma(const _Float16* __restrict__ A,
                             const _Float16* __restrict__ Bt,
                             float* __restrict__ z, int N, int tiles) {
    int wave = threadIdx.x >> 5;
    int lane = threadIdx.x & 31;
    int tile = blockIdx.x * 8 + wave;
    if (tile >= tiles) return;               // wave-uniform: EXEC stays all-ones
    int m = lane & 15, grp = lane >> 4;
    int row = tile * 16 + m;
    if (row >= N) row = N - 1;               // branchless clamp (stores guarded)
    const _Float16* Arow = A + (long)row * 256;
    const _Float16* Brow = Bt + m * 256;
    f8 c = {};
    #pragma unroll
    for (int kb = 0; kb < 8; ++kb) {
        h16 a, b;
        #pragma unroll
        for (int v = 0; v < 8; ++v) {
            // A 16x32 f16 layout: VGPR v <-> K = grp*8 + (v<4 ? 2v : 16+2(v-4))
            int ka = kb * 32 + grp * 8 + ((v < 4) ? 2 * v : 16 + 2 * (v - 4));
            hp2 pa = *(const hp2*)(Arow + ka);
            a[2 * v] = pa[0]; a[2 * v + 1] = pa[1];
            // B 32x16 f16 layout: lane = N-col, VGPR v <-> K = grp*8 + (v<4 ? 2v : 16+2(v-4))
            int kbx = kb * 32 + grp * 8 + ((v < 4) ? 2 * v : 16 + 2 * (v - 4));
            hp2 pb = *(const hp2*)(Brow + kbx);
            b[2 * v] = pb[0]; b[2 * v + 1] = pb[1];
        }
        c = __builtin_amdgcn_wmma_f32_16x16x32_f16(false, a, false, b,
                                                   (short)0, c, false, false);
    }
    #pragma unroll
    for (int r = 0; r < 8; ++r) {            // D: row = r + 8*grp, col = lane&15
        int mr = tile * 16 + r + 8 * grp;
        if (mr < N) z[(long)mr * 16 + m] = c[r];
    }
}

__global__ void k_coef2(const float* __restrict__ z,
                        const float* __restrict__ as2, const float* __restrict__ ad2,
                        float* __restrict__ aS, float* __restrict__ aD, int N) {
    int n = blockIdx.x * blockDim.x + threadIdx.x;
    if (n >= N) return;
    float s = 0.f, d = 0.f;
    #pragma unroll
    for (int c = 0; c < 16; ++c) {
        float v = z[(long)n * 16 + c];
        s += v * as2[c];
        d += v * ad2[c];
    }
    aS[n] = s; aD[n] = d;
}

__global__ void k_init2(float* __restrict__ out, unsigned* __restrict__ emax,
                        float* __restrict__ denom, const float* __restrict__ b2,
                        int NC, int N) {
    int i = blockIdx.x * blockDim.x + threadIdx.x;
    if (i < NC) out[i] = b2[i & 15];
    if (i < N) { emax[i] = ORD_NEG_INF; denom[i] = 0.f; }
}

__global__ void k_edge_max2(const int* __restrict__ ei, int E,
                            const float* __restrict__ aS, const float* __restrict__ aD,
                            unsigned* __restrict__ emax, int EE) {
    int e = blockIdx.x * blockDim.x + threadIdx.x;
    if (e >= EE) return;
    int s, d; edge_nodes(ei, E, e, s, d);
    float sc = aS[s] + aD[d];
    sc = sc > 0.f ? sc : NEG_SLOPE * sc;
    atomicMax(&emax[d], ford(sc));
}

__global__ void k_edge_sum2(const int* __restrict__ ei, int E,
                            const float* __restrict__ aS, const float* __restrict__ aD,
                            const unsigned* __restrict__ emax,
                            float* __restrict__ denom, int EE) {
    int e = blockIdx.x * blockDim.x + threadIdx.x;
    if (e >= EE) return;
    int s, d; edge_nodes(ei, E, e, s, d);
    float sc = aS[s] + aD[d];
    sc = sc > 0.f ? sc : NEG_SLOPE * sc;
    atomF(&denom[d], __expf(sc - funord(emax[d])));
}

__global__ void k_edge_agg2(const int* __restrict__ ei, int E,
                            const float* __restrict__ aS, const float* __restrict__ aD,
                            const unsigned* __restrict__ emax,
                            const float* __restrict__ denom,
                            const float* __restrict__ z, float* __restrict__ out,
                            int EE) {
    int gid = blockIdx.x * blockDim.x + threadIdx.x;
    if (gid >= EE * 16) return;
    int e = gid >> 4, c = gid & 15, s, d;
    edge_nodes(ei, E, e, s, d);
    float sc = aS[s] + aD[d];
    sc = sc > 0.f ? sc : NEG_SLOPE * sc;
    float alpha = __expf(sc - funord(emax[d])) / (denom[d] + 1e-16f);
    atomF(&out[(long)d * 16 + c], z[(long)s * 16 + c] * alpha);
}

extern "C" void kernel_launch(void* const* d_in, const int* in_sizes, int n_in,
                              void* d_out, int out_size, void* d_ws, size_t ws_size,
                              hipStream_t stream) {
    const float* x   = (const float*)d_in[0];
    const int*   ei  = (const int*)  d_in[1];
    const float* W1  = (const float*)d_in[2];
    const float* as1 = (const float*)d_in[3];
    const float* ad1 = (const float*)d_in[4];
    const float* b1  = (const float*)d_in[5];
    const float* W2  = (const float*)d_in[6];
    const float* as2 = (const float*)d_in[7];
    const float* ad2 = (const float*)d_in[8];
    const float* b2  = (const float*)d_in[9];

    const int N  = in_sizes[0] / 3;
    const int E  = in_sizes[1] / 2;
    const int EE = E + N;                 // self loops appended

    char* w = (char*)d_ws;
    auto alloc = [&](size_t bytes) -> char* {
        char* p = w;
        w += (bytes + 255) & ~(size_t)255;
        return p;
    };
    float*     h1  = (float*)    alloc((size_t)N * 256 * 4);
    float*     out1= (float*)    alloc((size_t)N * 256 * 4);
    _Float16*  hh  = (_Float16*) alloc((size_t)N * 256 * 2);
    float*     z   = (float*)    alloc((size_t)N * 16 * 4);
    float*     aS1 = (float*)    alloc((size_t)N * 8 * 4);
    float*     aD1 = (float*)    alloc((size_t)N * 8 * 4);
    unsigned*  em1 = (unsigned*) alloc((size_t)N * 8 * 4);
    float*     dn1 = (float*)    alloc((size_t)N * 8 * 4);
    float*     aS2 = (float*)    alloc((size_t)N * 4);
    float*     aD2 = (float*)    alloc((size_t)N * 4);
    unsigned*  em2 = (unsigned*) alloc((size_t)N * 4);
    float*     dn2 = (float*)    alloc((size_t)N * 4);
    _Float16*  W2t = (_Float16*) alloc(4096 * 2);

    float* out = (float*)d_out;
    const int NC1 = N * 256, NH1 = N * 8, NC2 = N * 16;
    const int tiles = (N + 15) / 16;

    // ---- Layer 1 ----
    k_layer1_node<<<N, 256, 0, stream>>>(x, W1, as1, ad1, h1, aS1, aD1);
    k_init1<<<(NC1 + 255) / 256, 256, 0, stream>>>(out1, em1, dn1, b1, NC1, NH1);
    k_edge_max1<<<(EE * 8 + 255) / 256, 256, 0, stream>>>(ei, E, aS1, aD1, em1, EE);
    k_edge_sum1<<<(EE * 8 + 255) / 256, 256, 0, stream>>>(ei, E, aS1, aD1, em1, dn1, EE);
    k_edge_agg1<<<EE, 256, 0, stream>>>(ei, E, aS1, aD1, em1, dn1, h1, out1);
    k_elu_tohalf<<<(NC1 + 255) / 256, 256, 0, stream>>>(out1, hh, NC1);

    // ---- Layer 2 ----
    k_w2t<<<16, 256, 0, stream>>>(W2, W2t);
    k_gemm2_wmma<<<(tiles + 7) / 8, 256, 0, stream>>>(hh, W2t, z, N, tiles);
    k_coef2<<<(N + 255) / 256, 256, 0, stream>>>(z, as2, ad2, aS2, aD2, N);
    k_init2<<<(NC2 + 255) / 256, 256, 0, stream>>>(out, em2, dn2, b2, NC2, N);
    k_edge_max2<<<(EE + 255) / 256, 256, 0, stream>>>(ei, E, aS2, aD2, em2, EE);
    k_edge_sum2<<<(EE + 255) / 256, 256, 0, stream>>>(ei, E, aS2, aD2, em2, dn2, EE);
    k_edge_agg2<<<(EE * 16 + 255) / 256, 256, 0, stream>>>(ei, E, aS2, aD2, em2, dn2, z, out, EE);
}
